// CameraViewTransformerLSS_81372450390068
// MI455X (gfx1250) — compile-verified, matrix operand order checked by reading, not run on hardware
//
#include <hip/hip_runtime.h>
#include <hip/hip_bf16.h>

typedef __attribute__((ext_vector_type(16))) _Float16 v16h;
typedef __attribute__((ext_vector_type(8)))  float    v8f;

#define NCAM 6
#define NDEP 48
#define HF   64
#define WF   176
#define CCTX 80
#define BEVH 128
#define BEVW 128
#define BEVC 128
#define HWF  (HF * WF)          // 11264
#define NPIX (NCAM * HWF)       // 67584
#define HWBEV (BEVH * BEVW)     // 16384
#define NBEV (HWBEV * CCTX)     // 1310720 floats

// ---------------------------------------------------------------------------
// 0) zero the BEV accumulator
// ---------------------------------------------------------------------------
__global__ void zero_bev_kernel(float* __restrict__ bev) {
    int i = blockIdx.x * blockDim.x + threadIdx.x;
    if (i < NBEV) bev[i] = 0.0f;
}

// ---------------------------------------------------------------------------
// 1) transpose context [n, c, h*w] -> ctx_t [n*h*w, c]   (LDS tiled)
// ---------------------------------------------------------------------------
__global__ void transpose_ctx_kernel(const float* __restrict__ ctx,
                                     float* __restrict__ ctx_t) {
    __shared__ float tile[32][33];
    const int n  = blockIdx.z;
    const int p0 = blockIdx.x * 32;   // pixel-within-camera tile base (HWF % 32 == 0)
    const int c0 = blockIdx.y * 32;   // channel tile base
    const int tx = threadIdx.x;       // 0..31
    const int ty = threadIdx.y;       // 0..7

#pragma unroll
    for (int i = 0; i < 4; ++i) {
        int c = c0 + ty + i * 8;
        if (c < CCTX)
            tile[ty + i * 8][tx] = ctx[(n * CCTX + c) * HWF + p0 + tx];
    }
    __syncthreads();
#pragma unroll
    for (int i = 0; i < 4; ++i) {
        int p = p0 + ty + i * 8;
        int c = c0 + tx;
        if (c < CCTX)
            ctx_t[(size_t)(n * HWF + p) * CCTX + c] = tile[tx][ty + i * 8];
    }
}

// ---------------------------------------------------------------------------
// 2) splat: one wave (32 lanes) per pixel; loop depth bins, atomicAdd 80 chans
// ---------------------------------------------------------------------------
__global__ __launch_bounds__(256)
void splat_kernel(const float* __restrict__ depth_prob,
                  const float* __restrict__ ctx_t,
                  const float* __restrict__ intr,
                  const float* __restrict__ c2e,
                  float* __restrict__ bev) {
    const int wave  = (blockIdx.x * blockDim.x + threadIdx.x) >> 5;
    const int lane  = threadIdx.x & 31;
    if (wave >= NPIX) return;

    const int n   = wave / HWF;
    const int rem = wave - n * HWF;
    const int h   = rem / WF;
    const int w   = rem - h * WF;

    // per-pixel geometry constants
    const float fx = intr[n * 9 + 0];
    const float fy = intr[n * 9 + 4];
    const float cx = intr[n * 9 + 2];
    const float cy = intr[n * 9 + 5];
    const float u  = ((float)w + 0.5f) * 4.0f;
    const float v  = ((float)h + 0.5f) * 4.0f;
    const float ax = (u - cx) / fx;
    const float ay = (v - cy) / fy;

    const float* T = c2e + n * 16;
    const float px = T[0] * ax + T[1] * ay + T[2];
    const float tx = T[3];
    const float py = T[4] * ax + T[5] * ay + T[6];
    const float ty = T[7];

    const float inv_mx = 1.0f / (100.0f / (float)BEVW);  // = 1.28
    const float inv_my = 1.0f / (100.0f / (float)BEVH);

    // context channels for this pixel (coalesced from transposed layout)
    const float* cp = ctx_t + (size_t)wave * CCTX;
    const float ctx0 = cp[lane];
    const float ctx1 = cp[32 + lane];
    const float ctx2 = (lane < 16) ? cp[64 + lane] : 0.0f;

    // depth probs: lane l holds d=l; lanes 0..15 also hold d=32+l
    const float* dpb = depth_prob + ((size_t)(n * NDEP) * HF + h) * WF + w;
    const float dp_a = dpb[(size_t)lane * HWF];
    const float dp_b = (lane < 16) ? dpb[(size_t)(32 + lane) * HWF] : 0.0f;

    const float zstep = (60.0f - 1.0f) / (float)(NDEP - 1);

    for (int d = 0; d < NDEP; ++d) {
        const float dp = (d < 32) ? __shfl(dp_a, d, 32) : __shfl(dp_b, d - 32, 32);
        const float Z  = 1.0f + zstep * (float)d;
        const float xe = Z * px + tx;
        const float ye = Z * py + ty;
        const int ix = (int)floorf((xe + 50.0f) * inv_mx);
        const int iy = (int)floorf((ye + 50.0f) * inv_my);
        if (ix >= 0 && ix < BEVW && iy >= 0 && iy < BEVH) {
            float* b = bev + (size_t)(iy * BEVW + ix) * CCTX;
            atomicAdd(b + lane,      dp * ctx0);
            atomicAdd(b + 32 + lane, dp * ctx1);
            if (lane < 16) atomicAdd(b + 64 + lane, dp * ctx2);
        }
    }
}

// ---------------------------------------------------------------------------
// 3) projection GEMM (WMMA f16->f32) + bias + BN + ReLU
//    y[o, pos] = relu( (sum_c bev[pos,c] * w[o,c] + b[o] - mean[o]) * inv[o] + beta[o] )
//    one wave per 16(pos) x 16(o) tile; K = 80 padded to 96 (3 x K32 WMMA)
// ---------------------------------------------------------------------------
__global__ __launch_bounds__(256)
void proj_bn_relu_kernel(const float* __restrict__ bev,
                         const float* __restrict__ w_proj,
                         const float* __restrict__ b_proj,
                         const float* __restrict__ gamma,
                         const float* __restrict__ beta,
                         const float* __restrict__ mean,
                         const float* __restrict__ var,
                         float* __restrict__ out) {
    const int pos_base = blockIdx.x * 16;           // 1024 blocks cover 16384 pos
    const int o_tile   = threadIdx.x >> 5;          // 8 waves -> 8 o-tiles (128 ch)
    const int o_base   = o_tile * 16;
    const int lane     = threadIdx.x & 31;
    const int half     = lane >> 4;
    const int r        = lane & 15;

    v8f acc = {};
#pragma unroll
    for (int kc = 0; kc < 3; ++kc) {
        const int c0 = kc * 32;
        v16h a, b;
#pragma unroll
        for (int e = 0; e < 16; ++e) {
            // documented 16-bit 16x32 A layout: K(e) = e + 8*half (+8 if e>=8)
            const int k = c0 + ((e < 8) ? e : e + 8) + 8 * half;
            a[e] = (k < CCTX) ? (_Float16)bev[(size_t)(pos_base + r) * CCTX + k]
                              : (_Float16)0.0f;
            b[e] = (k < CCTX) ? (_Float16)w_proj[(size_t)(o_base + r) * CCTX + k]
                              : (_Float16)0.0f;
        }
        acc = __builtin_amdgcn_wmma_f32_16x16x32_f16(
            /*neg_a=*/false, a, /*neg_b=*/false, b,
            /*c_mod=*/(short)0, acc, /*reuse_a=*/false, /*reuse_b=*/false);
    }

    // epilogue: per-lane output channel o = o_base + r
    const int o    = o_base + r;
    const float bp = b_proj[o];
    const float mu = mean[o];
    const float iv = gamma[o] * rsqrtf(var[o] + 1e-5f);
    const float bt = beta[o];

#pragma unroll
    for (int i = 0; i < 8; ++i) {
        const int pos = pos_base + i + 8 * half;    // C/D layout: M = i (+8 for hi lanes)
        float y = (acc[i] + bp - mu) * iv + bt;
        y = fmaxf(y, 0.0f);
        out[(size_t)o * HWBEV + pos] = y;
    }
}

// ---------------------------------------------------------------------------
extern "C" void kernel_launch(void* const* d_in, const int* in_sizes, int n_in,
                              void* d_out, int out_size, void* d_ws, size_t ws_size,
                              hipStream_t stream) {
    const float* depth_prob = (const float*)d_in[0];
    const float* context    = (const float*)d_in[1];
    const float* intrinsics = (const float*)d_in[2];
    const float* cam2ego    = (const float*)d_in[3];
    const float* w_proj     = (const float*)d_in[4];
    const float* b_proj     = (const float*)d_in[5];
    const float* bn_gamma   = (const float*)d_in[6];
    const float* bn_beta    = (const float*)d_in[7];
    const float* bn_mean    = (const float*)d_in[8];
    const float* bn_var     = (const float*)d_in[9];
    float* out = (float*)d_out;

    float* bev   = (float*)d_ws;                 // NBEV floats (5 MB)
    float* ctx_t = bev + NBEV;                   // NPIX*CCTX floats (21.6 MB)

    // 0) zero accumulator
    zero_bev_kernel<<<(NBEV + 255) / 256, 256, 0, stream>>>(bev);

    // 1) transpose context to [pixel, c]
    {
        dim3 grid(HWF / 32, (CCTX + 31) / 32, NCAM);
        dim3 block(32, 8);
        transpose_ctx_kernel<<<grid, block, 0, stream>>>(context, ctx_t);
    }

    // 2) splat (one wave per pixel; 8 waves per 256-thread block)
    {
        int blocks = (NPIX + 7) / 8;
        splat_kernel<<<blocks, 256, 0, stream>>>(depth_prob, ctx_t, intrinsics,
                                                 cam2ego, bev);
    }

    // 3) WMMA projection + BN + ReLU
    proj_bn_relu_kernel<<<HWBEV / 16, 256, 0, stream>>>(
        bev, w_proj, b_proj, bn_gamma, bn_beta, bn_mean, bn_var, out);
}
